// STCM_59072980189278
// MI455X (gfx1250) — matrix operand order repaired and checked by reading, not compiled, not run on hardware
//
#include <hip/hip_runtime.h>
#include <hip/hip_bf16.h>

typedef __attribute__((ext_vector_type(16))) _Float16 v16h;
typedef __attribute__((ext_vector_type(8)))  float    v8f;
typedef __attribute__((ext_vector_type(4)))  float    v4f;

// ---- fragment builders (load phase separated from convert phase so the
//      scheduler batches all loads before any cvt) -------------------------
__device__ __forceinline__ v16h frag_a_vec(const float* ap) {
    // ap points at A row start + akb; 16B-aligned in all ksA==1 call sites
    v4f a0 = *(const v4f*)(ap + 0);
    v4f a1 = *(const v4f*)(ap + 4);
    v4f a2 = *(const v4f*)(ap + 16);
    v4f a3 = *(const v4f*)(ap + 20);
    v16h f;
#pragma unroll
    for (int e = 0; e < 4; ++e) {
        f[e]      = (_Float16)a0[e];
        f[e + 4]  = (_Float16)a1[e];
        f[e + 8]  = (_Float16)a2[e];
        f[e + 12] = (_Float16)a3[e];
    }
    return f;
}

__device__ __forceinline__ v16h frag_a_strided(const float* ap, long long ksA,
                                               int akb) {
    float t[16];
#pragma unroll
    for (int e = 0; e < 16; ++e) {
        int kk = akb + ((e >> 3) << 4) + (e & 7);
        t[e] = ap[(long long)kk * ksA];
    }
    v16h f;
#pragma unroll
    for (int e = 0; e < 16; ++e) f[e] = (_Float16)t[e];
    return f;
}

__device__ __forceinline__ v16h frag_b(const float* bp, long long sBk) {
    float t[16];
#pragma unroll
    for (int e = 0; e < 16; ++e) t[e] = bp[(long long)(((e >> 3) << 4) + (e & 7)) * sBk];
    v16h f;
#pragma unroll
    for (int e = 0; e < 16; ++e) f[e] = (_Float16)t[e];
    return f;
}

__device__ __forceinline__ v16h frag_a_tail(const float* A, long long aoff,
                                            long long ksA, int akb, int K32,
                                            int K) {
    float t[16];
#pragma unroll
    for (int e = 0; e < 16; ++e) {
        int k  = K32 + akb + ((e >> 3) << 4) + (e & 7);
        int kc = (k < K) ? k : (K - 1);
        float msk = (k < K) ? 1.0f : 0.0f;
        t[e] = A[aoff + (long long)kc * ksA] * msk;
    }
    v16h f;
#pragma unroll
    for (int e = 0; e < 16; ++e) f[e] = (_Float16)t[e];
    return f;
}

__device__ __forceinline__ v16h frag_b_tail(const float* B, long long boff,
                                            long long sBk, int bkb, int K32,
                                            int K) {
    float t[16];
#pragma unroll
    for (int e = 0; e < 16; ++e) {
        int k  = K32 + bkb + ((e >> 3) << 4) + (e & 7);
        int kc = (k < K) ? k : (K - 1);
        float msk = (k < K) ? 1.0f : 0.0f;
        t[e] = B[(long long)kc * sBk + boff] * msk;
    }
    v16h f;
#pragma unroll
    for (int e = 0; e < 16; ++e) f[e] = (_Float16)t[e];
    return f;
}

// ---------------------------------------------------------------------------
// Generic batched-strided GEMM using V_WMMA_F32_16X16X32_F16.
//   A[b*bsA + r*rsA + k*ksA]   where m = b*rpb + r
//   B[k*sBk + n*sBn]
//   C[b*bsC + r*rsC + n*nsC]   (+ optional bias[n], relu, residual at C offset)
// One wave computes a 32x32 output macro-tile: 2 A-frags x 2 B-frags feed 4
// independent WMMAs per K-chunk (2x operand reuse, no D->A/B hazards).
// Out-of-range rows/cols handled purely by address clamping (their D values
// are never stored); only the K-tail is value-masked, by multiplication, so
// loads stay unconditional and pipeline without exec-divergence.
// ---------------------------------------------------------------------------
__global__ __launch_bounds__(32) void k_gemm_wmma(
    const float* __restrict__ A, const float* __restrict__ B,
    const float* __restrict__ bias, const float* __restrict__ resid,
    float* __restrict__ C,
    int M, int N, int K,
    long long bsA, long long rsA, long long ksA, int rpb,
    long long sBk, long long sBn,
    long long bsC, long long rsC, long long nsC,
    int relu)
{
    const int lane = threadIdx.x;      // 0..31
    const int lm   = lane & 15;
    const int hg   = lane >> 4;        // half-wave group
    const int rowBase = blockIdx.y * 32;
    const int colBase = blockIdx.x * 32;

    // A rows for the two M-subtiles (clamped in-bounds)
    const int am0  = rowBase + lm;
    const int am1  = am0 + 16;
    const int amc0 = (am0 < M) ? am0 : (M - 1);
    const int amc1 = (am1 < M) ? am1 : (M - 1);
    const int ab0 = amc0 / rpb, ar0 = amc0 - ab0 * rpb;
    const int ab1 = amc1 / rpb, ar1 = amc1 - ab1 * rpb;
    const long long aoff0 = (long long)ab0 * bsA + (long long)ar0 * rsA;
    const long long aoff1 = (long long)ab1 * bsA + (long long)ar1 * rsA;
    const int akb = hg * 8;

    // B cols for the two N-subtiles (clamped in-bounds)
    const int bn0  = colBase + lm;
    const int bn1  = bn0 + 16;
    const int bnc0 = (bn0 < N) ? bn0 : (N - 1);
    const int bnc1 = (bn1 < N) ? bn1 : (N - 1);
    const long long boff0 = (long long)bnc0 * sBn;
    const long long boff1 = (long long)bnc1 * sBn;
    const int bkb = hg * 8;

    const int K32 = K & ~31;           // full 32-wide chunks
    v8f acc00 = {}, acc01 = {}, acc10 = {}, acc11 = {};

    {
        const float* bp0 = B + boff0 + (long long)bkb * sBk;
        const float* bp1 = B + boff1 + (long long)bkb * sBk;
        if (ksA == 1) {
            const float* ap0 = A + aoff0 + akb;
            const float* ap1 = A + aoff1 + akb;
            for (int k0 = 0; k0 < K32; k0 += 32) {
                v16h af0 = frag_a_vec(ap0);
                v16h af1 = frag_a_vec(ap1);
                v16h bf0 = frag_b(bp0, sBk);
                v16h bf1 = frag_b(bp1, sBk);
                acc00 = __builtin_amdgcn_wmma_f32_16x16x32_f16(
                    false, af0, false, bf0, (short)0, acc00, false, false);
                acc01 = __builtin_amdgcn_wmma_f32_16x16x32_f16(
                    false, af0, false, bf1, (short)0, acc01, false, false);
                acc10 = __builtin_amdgcn_wmma_f32_16x16x32_f16(
                    false, af1, false, bf0, (short)0, acc10, false, false);
                acc11 = __builtin_amdgcn_wmma_f32_16x16x32_f16(
                    false, af1, false, bf1, (short)0, acc11, false, false);
                ap0 += 32; ap1 += 32;
                bp0 += 32 * sBk; bp1 += 32 * sBk;
            }
        } else {
            const float* ap0 = A + aoff0;
            const float* ap1 = A + aoff1;
            for (int k0 = 0; k0 < K32; k0 += 32) {
                v16h af0 = frag_a_strided(ap0, ksA, akb);
                v16h af1 = frag_a_strided(ap1, ksA, akb);
                v16h bf0 = frag_b(bp0, sBk);
                v16h bf1 = frag_b(bp1, sBk);
                acc00 = __builtin_amdgcn_wmma_f32_16x16x32_f16(
                    false, af0, false, bf0, (short)0, acc00, false, false);
                acc01 = __builtin_amdgcn_wmma_f32_16x16x32_f16(
                    false, af0, false, bf1, (short)0, acc01, false, false);
                acc10 = __builtin_amdgcn_wmma_f32_16x16x32_f16(
                    false, af1, false, bf0, (short)0, acc10, false, false);
                acc11 = __builtin_amdgcn_wmma_f32_16x16x32_f16(
                    false, af1, false, bf1, (short)0, acc11, false, false);
                ap0 += 32 * ksA; ap1 += 32 * ksA;
                bp0 += 32 * sBk; bp1 += 32 * sBk;
            }
        }
    }

    if (K32 < K) {
        v16h af0 = frag_a_tail(A, aoff0, ksA, akb, K32, K);
        v16h af1 = frag_a_tail(A, aoff1, ksA, akb, K32, K);
        v16h bf0 = frag_b_tail(B, boff0, sBk, bkb, K32, K);
        v16h bf1 = frag_b_tail(B, boff1, sBk, bkb, K32, K);
        acc00 = __builtin_amdgcn_wmma_f32_16x16x32_f16(
            false, af0, false, bf0, (short)0, acc00, false, false);
        acc01 = __builtin_amdgcn_wmma_f32_16x16x32_f16(
            false, af0, false, bf1, (short)0, acc01, false, false);
        acc10 = __builtin_amdgcn_wmma_f32_16x16x32_f16(
            false, af1, false, bf0, (short)0, acc10, false, false);
        acc11 = __builtin_amdgcn_wmma_f32_16x16x32_f16(
            false, af1, false, bf1, (short)0, acc11, false, false);
    }

    // Epilogue: subtile (ti,tj); lane holds rows m = rowBase+ti*16+hg*8+r,
    // col n = colBase + tj*16 + lm
    v8f accs[2][2] = { { acc00, acc01 }, { acc10, acc11 } };
#pragma unroll
    for (int ti = 0; ti < 2; ++ti) {
        const int m0 = rowBase + ti * 16 + (hg << 3);
        int cb = m0 / rpb;
        int cr = m0 - cb * rpb;
#pragma unroll
        for (int r = 0; r < 8; ++r) {
            int m = m0 + r;
            if (m < M) {
                long long cbase = (long long)cb * bsC + (long long)cr * rsC;
#pragma unroll
                for (int tj = 0; tj < 2; ++tj) {
                    int n = colBase + tj * 16 + lm;
                    if (n < N) {
                        float v = accs[ti][tj][r];
                        if (bias)  v += bias[n];
                        if (relu)  v = fmaxf(v, 0.0f);
                        long long coff = cbase + (long long)n * nsC;
                        if (resid) v += resid[coff];
                        C[coff] = v;
                    }
                }
            }
            if (++cr == rpb) { ++cb; cr = 0; }
        }
    }
}

// ---------------------------------------------------------------------------
// fe = relu(normalize(group_emb[gi] + off_emb@off_W + off_b)); eQ = fe@Wq + bq
// ---------------------------------------------------------------------------
__global__ __launch_bounds__(64) void k_fe_eq(
    const int* __restrict__ gidx, const float* __restrict__ gemb,
    const float* __restrict__ off_emb, const float* __restrict__ offW,
    const float* __restrict__ offb, const float* __restrict__ Wq,
    const float* __restrict__ bq, float* __restrict__ FE, float* __restrict__ EQ)
{
    __shared__ float sfe[64];
    __shared__ float red[64];
    int p = blockIdx.x, d = threadIdx.x;
    int g = gidx[p];
    float v = gemb[g * 64 + d] + offb[d];
    for (int k = 0; k < 10; ++k) v += off_emb[p * 10 + k] * offW[k * 64 + d];
    red[d] = v * v;
    __syncthreads();
    for (int s = 32; s > 0; s >>= 1) {
        if (d < s) red[d] += red[d + s];
        __syncthreads();
    }
    float nrm = fmaxf(sqrtf(red[0]), 1e-12f);
    float fe = fmaxf(v / nrm, 0.0f);
    sfe[d] = fe;
    FE[p * 64 + d] = fe;
    __syncthreads();
    if (d < 10) {
        float a = bq[d];
        for (int k = 0; k < 64; ++k) a += sfe[k] * Wq[k * 10 + d];
        EQ[p * 10 + d] = a;
    }
}

// A = softmax(eQ @ eQ.T, axis=0): one block per column j, 256 threads over i
__global__ __launch_bounds__(256) void k_colsoftmax_A(
    const float* __restrict__ EQ, float* __restrict__ Amat)
{
    __shared__ float qj[10];
    __shared__ float red[256];
    int j = blockIdx.x, i = threadIdx.x;
    if (i < 10) qj[i] = EQ[j * 10 + i];
    __syncthreads();
    float s = 0.f;
    for (int k = 0; k < 10; ++k) s += EQ[i * 10 + k] * qj[k];
    red[i] = s;
    __syncthreads();
    for (int w = 128; w > 0; w >>= 1) {
        if (i < w) red[i] = fmaxf(red[i], red[i + w]);
        __syncthreads();
    }
    float mx = red[0];
    __syncthreads();
    float e = expf(s - mx);
    red[i] = e;
    __syncthreads();
    for (int w = 128; w > 0; w >>= 1) {
        if (i < w) red[i] += red[i + w];
        __syncthreads();
    }
    Amat[i * 256 + j] = e / red[0];
}

// Q = Qf + interp(temp_emb)[t]; 52 blocks x 64 threads
__global__ __launch_bounds__(64) void k_qinterp(
    const float* __restrict__ QF, const float* __restrict__ temp_emb,
    const int* __restrict__ t, float* __restrict__ Q)
{
    int tt = blockIdx.x, d = threadIdx.x;
    int ti = t[tt];
    int i = ti / 7;
    float f = (float)(ti % 7) / 7.0f;
    int j = (i + 1 < 52) ? i + 1 : 52;
    Q[tt * 64 + d] = QF[tt * 64 + d] + temp_emb[i * 64 + d] * (1.0f - f) +
                     temp_emb[j * 64 + d] * f;
}

// att = softmax(Q @ Q.T / 8, axis=1): one block per row, 64 threads
__global__ __launch_bounds__(64) void k_att(
    const float* __restrict__ Q, float* __restrict__ ATT)
{
    __shared__ float qr[64];
    __shared__ float sc[64];
    int row = blockIdx.x, tid = threadIdx.x;
    qr[tid] = Q[row * 64 + tid];
    __syncthreads();
    float s = -1e30f;
    if (tid < 52) {
        float a = 0.f;
        for (int d = 0; d < 64; ++d) a += qr[d] * Q[tid * 64 + d];
        s = a * 0.125f;   // 1/sqrt(64)
    }
    sc[tid] = s;
    __syncthreads();
    for (int w = 32; w > 0; w >>= 1) {
        if (tid < w) sc[tid] = fmaxf(sc[tid], sc[tid + w]);
        __syncthreads();
    }
    float mx = sc[0];
    __syncthreads();
    float e = (tid < 52) ? expf(s - mx) : 0.f;
    sc[tid] = e;
    __syncthreads();
    for (int w = 32; w > 0; w >>= 1) {
        if (tid < w) sc[tid] += sc[tid + w];
        __syncthreads();
    }
    if (tid < 52) ATT[row * 52 + tid] = e / sc[0];
}

// SPART[tt,p,i] = (sum_c x4[tt,p,c]*CE[c,i]) * PE[p,i]; 1024 blocks x 256
__global__ __launch_bounds__(256) void k_s_part(
    const float* __restrict__ x4, const float* __restrict__ CE,
    const float* __restrict__ PE, float* __restrict__ SPART)
{
    __shared__ float red[256];
    int b = blockIdx.x;
    int tt = b >> 8, p = b & 255;
    int c = threadIdx.x;
    float xv = x4[(tt * 256 + p) * 256 + c];
    for (int i = 0; i < 10; ++i) {
        red[c] = xv * CE[c * 10 + i];
        __syncthreads();
        for (int w = 128; w > 0; w >>= 1) {
            if (c < w) red[c] += red[c + w];
            __syncthreads();
        }
        if (c == 0) SPART[b * 10 + i] = red[0] * PE[p * 10 + i];
        __syncthreads();
    }
}

// s3[i] = sum_{tt,p} SPART * TwE[tt,i]; feat = mE @ s3 ; single block, 512 thr
__global__ __launch_bounds__(512) void k_feat(
    const float* __restrict__ SPART, const float* __restrict__ TwE,
    const float* __restrict__ mE, float* __restrict__ FEAT)
{
    __shared__ float s3[10];
    int tid = threadIdx.x;
    if (tid < 10) {
        float a = 0.f;
        for (int idx = 0; idx < 1024; ++idx) {
            int tt = idx >> 8;
            a += SPART[idx * 10 + tid] * TwE[tt * 10 + tid];
        }
        s3[tid] = a;
    }
    __syncthreads();
    if (tid < 450) {
        float a = 0.f;
        for (int i = 0; i < 10; ++i) a += mE[tid * 10 + i] * s3[i];
        FEAT[tid] = a;
    }
}

// out[n] = act(bias[n] + sum_k in[k] * W[k*450+n]); 450x450 GEMV
__global__ __launch_bounds__(256) void k_gemv450(
    const float* __restrict__ in, const float* __restrict__ W,
    const float* __restrict__ bias, float* __restrict__ out, int relu)
{
    __shared__ float s_in[450];
    int tid = threadIdx.x;
    for (int k = tid; k < 450; k += 256) s_in[k] = in[k];
    __syncthreads();
    int col = blockIdx.x * 256 + tid;
    if (col < 450) {
        float a = bias[col];
        for (int k = 0; k < 450; ++k) a += s_in[k] * W[k * 450 + col];
        if (relu) a = fmaxf(a, 0.f);
        out[col] = a;
    }
}

// d_out += dec2_b + feat @ dec2_W   (the 472 MB streaming GEMV, NT b128 loads)
__global__ __launch_bounds__(256) void k_dec2(
    const float* __restrict__ feat, const float* __restrict__ W,
    const float* __restrict__ bias, float* __restrict__ out)
{
    __shared__ float sf[450];
    int tid = threadIdx.x;
    for (int k = tid; k < 450; k += 256) sf[k] = feat[k];
    __syncthreads();
    size_t col4 = (size_t)blockIdx.x * 256 + tid;   // 0..65535 (float4 columns)
    const v4f* W4 = (const v4f*)W;
    const v4f* B4 = (const v4f*)bias;
    v4f* O4 = (v4f*)out;
    v4f acc = B4[col4];
    for (int k = 0; k < 450; ++k) {
        float f = sf[k];
        v4f w = __builtin_nontemporal_load(&W4[(size_t)k * 65536 + col4]);
        acc += f * w;
    }
    O4[col4] = O4[col4] + acc;
}

// ---------------------------------------------------------------------------
static inline void gemm(hipStream_t s, const float* A, const float* B,
                        const float* bias, const float* resid, float* C,
                        int M, int N, int K,
                        long long bsA, long long rsA, long long ksA, int rpb,
                        long long sBk, long long sBn,
                        long long bsC, long long rsC, long long nsC, int relu)
{
    dim3 g((unsigned)((N + 31) / 32), (unsigned)((M + 31) / 32));
    k_gemm_wmma<<<g, 32, 0, s>>>(A, B, bias, resid, C, M, N, K,
                                 bsA, rsA, ksA, rpb, sBk, sBn,
                                 bsC, rsC, nsC, relu);
}

extern "C" void kernel_launch(void* const* d_in, const int* in_sizes, int n_in,
                              void* d_out, int out_size, void* d_ws, size_t ws_size,
                              hipStream_t stream)
{
    (void)in_sizes; (void)n_in; (void)out_size; (void)ws_size;

    const float* x        = (const float*)d_in[0];    // (52,256,256)
    const int*   t        = (const int*)  d_in[1];    // (52,)
    const int*   gidx     = (const int*)  d_in[2];    // (256,)
    const float* gemb     = (const float*)d_in[3];    // (16,64)
    const float* off_emb  = (const float*)d_in[4];    // (256,10)
    const float* off_W    = (const float*)d_in[5];    // (10,64)
    const float* off_b    = (const float*)d_in[6];
    const float* Wq       = (const float*)d_in[7];    // (64,10)
    const float* bq       = (const float*)d_in[8];
    // d_in[9]=Wk, d_in[10]=bk : unused in reference
    const float* city1_W  = (const float*)d_in[11];   // (256,256)
    const float* city1_b  = (const float*)d_in[12];
    const float* city2_W  = (const float*)d_in[13];   // (256,32)
    const float* city2_b  = (const float*)d_in[14];
    const float* pos1_W   = (const float*)d_in[15];   // (256,256)
    const float* pos1_b   = (const float*)d_in[16];
    const float* pos2_W   = (const float*)d_in[17];   // (256,32)
    const float* pos2_b   = (const float*)d_in[18];
    const float* temp_emb = (const float*)d_in[19];   // (53,64)
    const float* cp_W     = (const float*)d_in[20];   // (1024,64)
    const float* cp_b     = (const float*)d_in[21];
    const float* cityV1_W = (const float*)d_in[22];   // (256,32)
    const float* cityV1_b = (const float*)d_in[23];
    const float* cityV2_W = (const float*)d_in[24];   // (32,256)
    const float* cityV2_b = (const float*)d_in[25];
    const float* posV1_W  = (const float*)d_in[26];   // (256,32)
    const float* posV1_b  = (const float*)d_in[27];
    const float* posV2_W  = (const float*)d_in[28];   // (32,256)
    const float* posV2_b  = (const float*)d_in[29];
    const float* ffn1_W   = (const float*)d_in[30];   // (52,52)
    const float* ffn1_b   = (const float*)d_in[31];
    const float* ffn2_W   = (const float*)d_in[32];   // (52,52)
    const float* ffn2_b   = (const float*)d_in[33];
    const float* pred_W   = (const float*)d_in[34];   // (52,4)
    const float* pred_b   = (const float*)d_in[35];
    const float* TwE      = (const float*)d_in[36];   // (4,10)
    const float* PE       = (const float*)d_in[37];   // (256,10)
    const float* CE       = (const float*)d_in[38];   // (256,10)
    const float* mE       = (const float*)d_in[39];   // (450,10)
    const float* ex1_W    = (const float*)d_in[40];   // (450,450)
    const float* ex1_b    = (const float*)d_in[41];
    const float* ex2_W    = (const float*)d_in[42];
    const float* ex2_b    = (const float*)d_in[43];
    const float* dec1_W   = (const float*)d_in[44];
    const float* dec1_b   = (const float*)d_in[45];
    const float* dec2_W   = (const float*)d_in[46];   // (450,262144)
    const float* dec2_b   = (const float*)d_in[47];   // (262144,)

    float* ws = (float*)d_ws;
    float* out = (float*)d_out;                       // (4,256,256)

    // workspace layout (floats)
    const size_t OFF_FE  = 0;                         // 256*64
    const size_t OFF_EQ  = OFF_FE  + 256 * 64;        // 256*10
    const size_t OFF_AM  = OFF_EQ  + 256 * 10;        // 256*256
    const size_t OFF_QC  = OFF_AM  + 256 * 256;       // 52*256*32
    const size_t OFF_H2  = OFF_QC  + 52 * 256 * 32;   // 52*32*256
    const size_t OFF_QP  = OFF_H2  + 52 * 32 * 256;   // 52*32*32
    const size_t OFF_QF  = OFF_QP  + 52 * 32 * 32;    // 52*64
    const size_t OFF_Q   = OFF_QF  + 52 * 64;         // 52*64
    const size_t OFF_ATT = OFF_Q   + 52 * 64;         // 52*52
    const size_t OFF_T1  = OFF_ATT + 52 * 52;         // 13312*32
    const size_t OFF_T2  = OFF_T1  + 13312 * 32;      // 13312*32
    const size_t OFF_SP  = OFF_T2  + 13312 * 32;      // 1024*10
    const size_t OFF_FA  = OFF_SP  + 1024 * 10;       // 450
    const size_t OFF_FB  = OFF_FA  + 450;
    const size_t OFF_FC  = OFF_FB  + 450;
    const size_t OFF_FD  = OFF_FC  + 450;
    const size_t OFF_B0  = (OFF_FD + 450 + 255) & ~(size_t)255;  // 52*256*256
    const size_t OFF_B1  = OFF_B0 + (size_t)52 * 256 * 256;

    float* FE   = ws + OFF_FE;
    float* EQ   = ws + OFF_EQ;
    float* AM   = ws + OFF_AM;
    float* QC   = ws + OFF_QC;
    float* H2   = ws + OFF_H2;
    float* QP   = ws + OFF_QP;
    float* QF   = ws + OFF_QF;
    float* Qb   = ws + OFF_Q;
    float* ATT  = ws + OFF_ATT;
    float* T1   = ws + OFF_T1;
    float* T2   = ws + OFF_T2;
    float* SP   = ws + OFF_SP;
    float* FA   = ws + OFF_FA;
    float* FB   = ws + OFF_FB;
    float* FC   = ws + OFF_FC;
    float* FD   = ws + OFF_FD;
    float* B0   = ws + OFF_B0;     // ping
    float* B1   = ws + OFF_B1;     // pong

    // 1) fe + eQ
    k_fe_eq<<<256, 64, 0, stream>>>(gidx, gemb, off_emb, off_W, off_b, Wq, bq, FE, EQ);
    // 2) A = softmax(eQ eQ^T, axis=0)
    k_colsoftmax_A<<<256, 256, 0, stream>>>(EQ, AM);

    // 3) QX = x(13312x256) @ A(256x256)                       -> B0
    gemm(stream, x, AM, nullptr, nullptr, B0, 13312, 256, 256,
         (long long)13312 * 256, 256, 1, 13312, 256, 1,
         (long long)13312 * 256, 256, 1, 0);
    // 4) H = relu(QX @ city1_W + b)                           -> B1
    gemm(stream, B0, city1_W, city1_b, nullptr, B1, 13312, 256, 256,
         (long long)13312 * 256, 256, 1, 13312, 256, 1,
         (long long)13312 * 256, 256, 1, 1);
    // 5) QCpre = H @ city2_W + b    (52,256,32)               -> QC
    gemm(stream, B1, city2_W, city2_b, nullptr, QC, 13312, 32, 256,
         (long long)13312 * 256, 256, 1, 13312, 32, 1,
         (long long)13312 * 32, 32, 1, 0);
    // 6) H2 = relu(QC^T(0,2,1) @ pos1_W + b)  rows=(t,j), k=p -> H2 (52*32,256)
    gemm(stream, QC, pos1_W, pos1_b, nullptr, H2, 52 * 32, 256, 256,
         8192, 1, 32, 32, 256, 1,
         8192, 256, 1, 1);
    // 7) QP = H2 @ pos2_W + b   (1664,32)                     -> QP
    gemm(stream, H2, pos2_W, pos2_b, nullptr, QP, 52 * 32, 32, 256,
         (long long)1664 * 256, 256, 1, 1664, 32, 1,
         (long long)1664 * 32, 32, 1, 0);
    // 8) QF = QP.reshape(52,1024) @ cp_W + b   (52,64)        -> QF
    gemm(stream, QP, cp_W, cp_b, nullptr, QF, 52, 64, 1024,
         (long long)52 * 1024, 1024, 1, 52, 64, 1,
         (long long)52 * 64, 64, 1, 0);
    // 9) Q = QF + interp(temp_emb)[t]
    k_qinterp<<<52, 64, 0, stream>>>(QF, temp_emb, t, Qb);
    // 10) att = softmax(Q Q^T / 8, axis=1)
    k_att<<<52, 64, 0, stream>>>(Qb, ATT);

    // 11) T1 = relu(x @ cityV1_W + b)   (13312,32)            -> T1
    gemm(stream, x, cityV1_W, cityV1_b, nullptr, T1, 13312, 32, 256,
         (long long)13312 * 256, 256, 1, 13312, 32, 1,
         (long long)13312 * 32, 32, 1, 1);
    // 12) V1[t][c][p] = (T1 @ cityV2_W + b)[t,p,c]  transpose store -> B0
    gemm(stream, T1, cityV2_W, cityV2_b, nullptr, B0, 13312, 256, 32,
         8192, 32, 1, 256, 256, 1,
         65536, 1, 256, 0);
    // 13) T2 = relu(V1 @ posV1_W + b)   (13312,32)            -> T2
    gemm(stream, B0, posV1_W, posV1_b, nullptr, T2, 13312, 32, 256,
         (long long)13312 * 256, 256, 1, 13312, 32, 1,
         (long long)13312 * 32, 32, 1, 1);
    // 14) V2[t][p'][c] = (T2 @ posV2_W + b)[t,c,p']  transpose store -> B1
    gemm(stream, T2, posV2_W, posV2_b, nullptr, B1, 13312, 256, 32,
         8192, 32, 1, 256, 256, 1,
         65536, 1, 256, 0);
    // 15) VM = att(52x52) @ V2(52,65536) + x                  -> B0
    gemm(stream, ATT, B1, nullptr, x, B0, 52, 65536, 52,
         (long long)52 * 52, 52, 1, 52, 65536, 1,
         (long long)52 * 65536, 65536, 1, 0);
    // 16) VT1 = relu(VM^T(2,1,0)(65536,52) @ ffn1 + b)        -> B1
    gemm(stream, B0, ffn1_W, ffn1_b, nullptr, B1, 65536, 52, 52,
         1, 256, 65536, 256, 52, 1,
         (long long)256 * 52, 52, 1, 1);
    // 17) VT2 = VT1 @ ffn2 + b                                -> B0
    gemm(stream, B1, ffn2_W, ffn2_b, nullptr, B0, 65536, 52, 52,
         (long long)256 * 52, 52, 1, 256, 52, 1,
         (long long)256 * 52, 52, 1, 0);
    // 18) out_main = (VT2 @ pred_W + b) transposed (2,1,0)    -> d_out
    gemm(stream, B0, pred_W, pred_b, nullptr, out, 65536, 4, 52,
         (long long)256 * 52, 52, 1, 256, 4, 1,
         1, 256, 65536, 0);

    // 19-21) extra path: x4 contraction -> feat -> 450-MLP chain
    k_s_part<<<1024, 256, 0, stream>>>(x + (size_t)48 * 256 * 256, CE, PE, SP);
    k_feat<<<1, 512, 0, stream>>>(SP, TwE, mE, FA);
    k_gemv450<<<2, 256, 0, stream>>>(FA, ex1_W, ex1_b, FB, 1);
    k_gemv450<<<2, 256, 0, stream>>>(FB, ex2_W, ex2_b, FC, 0);
    k_gemv450<<<2, 256, 0, stream>>>(FC, dec1_W, dec1_b, FD, 1);
    // 22) d_out += dec2_b + FD @ dec2_W   (streaming 472 MB, NT loads)
    k_dec2<<<256, 256, 0, stream>>>(FD, dec2_W, dec2_b, out);
}
